// AssociationBinaryGPT_66838281060908
// MI455X (gfx1250) — matrix operand order repaired and checked by reading, not compiled, not run on hardware
//
#include <hip/hip_runtime.h>
#include <hip/hip_bf16.h>
#include <math.h>

typedef __attribute__((ext_vector_type(16))) _Float16 v16h;
typedef __attribute__((ext_vector_type(8)))  float    v8f;

#define NTHR 128
#define TILE_M 16
#define BIGM (1 << 30)

constexpr int LD_CAT = 584;   // concat / wide activation buffer (<=576 cols)
constexpr int LD_SC  = 392;   // scores / mid buffer (<=384 cols)
constexpr int LD_V   = 200;   // 192-wide vectors
constexpr float RSCALE = 0.07216878364870323f; // 1/sqrt(192)

// ---------------- packed-weight offsets (in _Float16 elements) ----------------
constexpr int pkhalves(int N, int K) { return N * ((K + 31) & ~31); }
constexpr int OFF_C2   = 0;
constexpr int OFF_C3   = OFF_C2   + pkhalves(96, 384);
constexpr int OFF_C4   = OFF_C3   + pkhalves(96, 384);
constexpr int OFF_PDP  = OFF_C4   + pkhalves(96, 576);
constexpr int OFF_E1   = OFF_PDP  + pkhalves(192, 288);
constexpr int OFF_E2   = OFF_E1   + pkhalves(192, 384);
constexpr int OFF_MEMK = OFF_E2   + pkhalves(192, 192);     // 3 hops contiguous
constexpr int OFF_MEMV = OFF_MEMK + 3 * pkhalves(384, 192);
constexpr int OFF_RL0  = OFF_MEMV + 3 * pkhalves(192, 384);
constexpr int OFF_RL1  = OFF_RL0  + pkhalves(192, 384);
constexpr int OFF_HQ   = OFF_RL1  + pkhalves(192, 384);
constexpr int OFF_HPT  = OFF_HQ   + pkhalves(192, 192);     // hq @ patterns^T
constexpr int OFF_HPP  = OFF_HPT  + pkhalves(256, 192);     // attn @ patterns
constexpr int OFF_HV   = OFF_HPP  + pkhalves(192, 256);
constexpr int OFF_DMK  = OFF_HV   + pkhalves(192, 192);
constexpr int OFF_DMV  = OFF_DMK  + pkhalves(384, 192);
constexpr int OFF_FUSE = OFF_DMV  + pkhalves(192, 384);
constexpr int OFF_H1   = OFF_FUSE + pkhalves(192, 576);
constexpr int OFF_H2   = OFF_H1   + pkhalves(384, 247);
constexpr int PK_TOTAL = OFF_H2   + pkhalves(192, 384);

__device__ __forceinline__ float gelu_f(float x) {
    return 0.5f * x * (1.0f + erff(x * 0.7071067811865476f));
}

// ---------------- weight pre-pack into WMMA B-fragment layout ----------------
// Packed order: halves[((nt*ktiles + kt)*32 + lane)*16 + i]
// B(n,k) = W[n*ldn + (k%KM)*S1 + (k/KM)*S2], zero for k>=Ktot.
__global__ __launch_bounds__(256) void pack_w(
    const float* __restrict__ W, _Float16* __restrict__ PK,
    int Ntot, int Ktot, int ldn, int KM, int S1, int S2)
{
    int Kpad = (Ktot + 31) & ~31;
    int ktiles = Kpad >> 5;
    int total = Ntot * Kpad;
    int idx = blockIdx.x * 256 + threadIdx.x;
    if (idx >= total) return;
    int i    = idx & 15;
    int lane = (idx >> 4) & 31;
    int tile = idx >> 9;
    int kt = tile % ktiles, nt = tile / ktiles;
    int n  = (nt << 4) + (lane & 15);
    int kb = (kt << 5) + ((lane >> 4) << 4) + i;
    float v = 0.f;
    if (kb < Ktot) v = W[n * ldn + (kb % KM) * S1 + (kb / KM) * S2];
    PK[idx] = (_Float16)v;
}

// ---------------- packed-weight WMMA GEMM (fast path) ----------------
// Software-pipelined: next k-tile's B fragment is issued before the current
// WMMA consumes its operand, so the wait before each WMMA is loadcnt<=2
// instead of 0 and L2 latency overlaps the matrix op.
template<int NT16, int KT32>
__device__ __forceinline__ void wmma_gemm_pk(
    const _Float16* __restrict__ PK, const float* __restrict__ bias,
    const float* actL, int lda, float* outL, int ldo, int ocol0)
{
    const int lane = threadIdx.x & 31;
    const int wv   = threadIdx.x >> 5;
    const int m    = lane & 15;
    const int g    = lane >> 4;
    const float* ap0 = actL + m * lda + (g << 4);
    for (int nt = wv; nt < NT16; nt += 4) {
        v8f c = {0.f, 0.f, 0.f, 0.f, 0.f, 0.f, 0.f, 0.f};
        const v16h* bp = (const v16h*)PK + nt * KT32 * 32 + lane;
        v16h bcur = bp[0];
#pragma unroll
        for (int kt = 0; kt < KT32; ++kt) {
            v16h bnxt = bcur;
            if (kt + 1 < KT32) bnxt = bp[(kt + 1) * 32];   // prefetch next tile
            const float* ap = ap0 + kt * 32;
            float4 p0 = *(const float4*)(ap);
            float4 p1 = *(const float4*)(ap + 4);
            float4 p2 = *(const float4*)(ap + 8);
            float4 p3 = *(const float4*)(ap + 12);
            v16h a;
            a[0]=(_Float16)p0.x;  a[1]=(_Float16)p0.y;  a[2]=(_Float16)p0.z;  a[3]=(_Float16)p0.w;
            a[4]=(_Float16)p1.x;  a[5]=(_Float16)p1.y;  a[6]=(_Float16)p1.z;  a[7]=(_Float16)p1.w;
            a[8]=(_Float16)p2.x;  a[9]=(_Float16)p2.y;  a[10]=(_Float16)p2.z; a[11]=(_Float16)p2.w;
            a[12]=(_Float16)p3.x; a[13]=(_Float16)p3.y; a[14]=(_Float16)p3.z; a[15]=(_Float16)p3.w;
            c = __builtin_amdgcn_wmma_f32_16x16x32_f16(
                    false, a, false, bcur, (short)0, c, false, false);
            bcur = bnxt;
        }
        float bv = bias ? bias[(nt << 4) + m] : 0.f;
#pragma unroll
        for (int r = 0; r < 8; ++r)
            outL[(r + (g << 3)) * ldo + ocol0 + (nt << 4) + m] = c[r] + bv;
    }
}

// ---------------- direct-read WMMA GEMM (fallback if ws too small) ----------------
__device__ __forceinline__ void wmma_gemm(
    const float* __restrict__ W, const float* __restrict__ bias,
    const float* actL, int lda,
    float* outL, int ldo, int ocol0,
    int Ntot, int Ktot, int ldn, int KM, int S1, int S2)
{
    const int lane = threadIdx.x & 31;
    const int wv   = threadIdx.x >> 5;
    const int m    = lane & 15;
    const int g    = lane >> 4;
    const int Kpad = (Ktot + 31) & ~31;
    const int ntiles = Ntot >> 4;
    for (int nt = wv; nt < ntiles; nt += 4) {
        const int n = (nt << 4) + m;
        v8f c = {0.f, 0.f, 0.f, 0.f, 0.f, 0.f, 0.f, 0.f};
        for (int k0 = 0; k0 < Kpad; k0 += 32) {
            v16h a, b;
#pragma unroll
            for (int i = 0; i < 16; ++i) {
                int ka = k0 + (g << 4) + i;
                a[i] = (_Float16)actL[m * lda + ka];
                int kb = k0 + (g << 4) + i;
                float wval = 0.f;
                if (kb < Ktot)
                    wval = W[n * ldn + (kb % KM) * S1 + (kb / KM) * S2];
                b[i] = (_Float16)wval;
            }
            c = __builtin_amdgcn_wmma_f32_16x16x32_f16(
                    false, a, false, b, (short)0, c, false, false);
        }
        float bv = bias ? bias[n] : 0.f;
#pragma unroll
        for (int r = 0; r < 8; ++r)
            outL[(r + (g << 3)) * ldo + ocol0 + n] = c[r] + bv;
    }
}

// 16 rows x d cols, 8 lanes per row (NTHR=128)
__device__ __forceinline__ void softmax_rows(float* buf, int ld, int d, float scale) {
    int row = threadIdx.x >> 3, sub = threadIdx.x & 7;
    float mx = -1e30f;
    for (int j = sub; j < d; j += 8) mx = fmaxf(mx, buf[row * ld + j] * scale);
    for (int o = 4; o; o >>= 1) mx = fmaxf(mx, __shfl_xor(mx, o, 8));
    float sm = 0.f;
    for (int j = sub; j < d; j += 8) {
        float e = __expf(buf[row * ld + j] * scale - mx);
        buf[row * ld + j] = e;
        sm += e;
    }
    for (int o = 4; o; o >>= 1) sm += __shfl_xor(sm, o, 8);
    float inv = 1.f / sm;
    for (int j = sub; j < d; j += 8) buf[row * ld + j] *= inv;
}

__device__ __forceinline__ void layernorm_rows(float* buf, int ld, int d,
                                               const float* gg, const float* bb) {
    int row = threadIdx.x >> 3, sub = threadIdx.x & 7;
    float s = 0.f, s2 = 0.f;
    for (int j = sub; j < d; j += 8) {
        float v = buf[row * ld + j]; s += v; s2 += v * v;
    }
    for (int o = 4; o; o >>= 1) { s += __shfl_xor(s, o, 8); s2 += __shfl_xor(s2, o, 8); }
    float mean = s / d;
    float var  = s2 / d - mean * mean;
    float inv  = rsqrtf(var + 1e-5f);
    for (int j = sub; j < d; j += 8)
        buf[row * ld + j] = (buf[row * ld + j] - mean) * inv * gg[j] + bb[j];
}

#define FORALL(idx, n) for (int idx = threadIdx.x; idx < (n); idx += NTHR)

// Unified call site: packed fast path or direct fallback
#define GEMM(Ntot, Ktot, PKOFF, W, BIAS, ACT, LDA, OUT, LDO, OC, ldn, KM, S1, S2)          \
    do { if constexpr (P)                                                                   \
            wmma_gemm_pk<(Ntot)/16, (((Ktot)+31)&~31)/32>(PKW + (PKOFF), BIAS, ACT, LDA,   \
                                                          OUT, LDO, OC);                    \
         else wmma_gemm(W, BIAS, ACT, LDA, OUT, LDO, OC, Ntot, Ktot, ldn, KM, S1, S2);      \
    } while (0)

template<bool P>
__global__ __launch_bounds__(NTHR) void abgpt_fused(
    const int* __restrict__ ctx_bytes, const int* __restrict__ bit_pos,
    const float* __restrict__ partial, const float* __restrict__ instinct,
    const float* byte_embed, const float* pos_embed,
    const float* c2w, const float* c2b, const float* c3w, const float* c3b,
    const float* c4w, const float* c4b,
    const float* pdn_g, const float* pdn_b, const float* pdp_w, const float* pdp_b,
    const float* e1w, const float* e1b, const float* n1g, const float* n1b,
    const float* e2w, const float* e2b, const float* n2g, const float* n2b,
    const float* mem_keys, const float* mem_vals,
    const float* rl0w, const float* rl0b, const float* rl1w, const float* rl1b,
    const float* rn0g, const float* rn0b, const float* rn1g, const float* rn1b,
    const float* hop_pat, const float* hqw, const float* hqb,
    const float* hvw, const float* hvb, const float* hng, const float* hnb,
    const float* dmk, const float* dmv,
    const float* gw, const float* gb, const float* fw, const float* fb,
    const float* bit_embed,
    const float* h1w, const float* h1b, const float* hn1g, const float* hn1b,
    const float* h2w, const float* h2b, const float* h3w, const float* h3b,
    const _Float16* __restrict__ PKW,
    float* __restrict__ out)
{
    __shared__ alignas(16) float CAT[TILE_M * LD_CAT];
    __shared__ alignas(16) float SC [TILE_M * LD_SC];
    __shared__ alignas(16) float XL [TILE_M * LD_V];
    __shared__ alignas(16) float Q  [TILE_M * LD_V];
    __shared__ alignas(16) float CUR[TILE_M * LD_V];
    __shared__ alignas(16) float TOT[TILE_M * LD_V];
    __shared__ alignas(16) float R  [TILE_M * LD_V];
    __shared__ alignas(16) float HOP[TILE_M * LD_V];   // also used as patterns_last
    __shared__ alignas(16) float DIR[TILE_M * LD_V];
    __shared__ float GT [TILE_M * 4];

    const int b0 = blockIdx.x * TILE_M;

    // ---- Stage 0: U = [x61 | x62 | x63], x_t = byte_embed[ctx[t]] + pos_embed[t]
    FORALL(idx, TILE_M * 576) {
        int m = idx / 576, c = idx % 576;
        int t = 61 + c / 192, e = c % 192;
        int by = ctx_bytes[(b0 + m) * 64 + t];
        float v = byte_embed[by * 192 + e] + pos_embed[t * 192 + e];
        CAT[m * LD_CAT + c] = v;
        if (t == 63) XL[m * LD_V + e] = v;
    }
    __syncthreads();

    // ---- Stage 1: convs at last position (tap layout via KM=192 mapping)
    GEMM(96, 384, OFF_C2, c2w, c2b, &CAT[192], LD_CAT, SC, LD_SC, 0,   384, 192, 2, 1);
    GEMM(96, 384, OFF_C3, c3w, c3b, &CAT[192], LD_CAT, SC, LD_SC, 96,  576, 192, 3, 1);
    GEMM(96, 576, OFF_C4, c4w, c4b, &CAT[0],   LD_CAT, SC, LD_SC, 192, 768, 192, 4, 1);
    __syncthreads();
    FORALL(idx, TILE_M * 288) { int m = idx / 288, c = idx % 288;
        SC[m * LD_SC + c] = gelu_f(SC[m * LD_SC + c]); }
    __syncthreads();
    layernorm_rows(SC, LD_SC, 288, pdn_g, pdn_b);
    __syncthreads();
    GEMM(192, 288, OFF_PDP, pdp_w, pdp_b, SC, LD_SC, HOP, LD_V, 0, 288, BIGM, 1, 0);
    __syncthreads();

    // ---- Stage 2: encoder -> query
    FORALL(idx, TILE_M * 384) { int m = idx / 384, c = idx % 384;
        CAT[m * LD_CAT + c] = (c < 192) ? XL[m * LD_V + c] : HOP[m * LD_V + (c - 192)]; }
    __syncthreads();
    GEMM(192, 384, OFF_E1, e1w, e1b, CAT, LD_CAT, CUR, LD_V, 0, 384, BIGM, 1, 0);
    __syncthreads();
    layernorm_rows(CUR, LD_V, 192, n1g, n1b);
    __syncthreads();
    FORALL(idx, TILE_M * 192) { int m = idx / 192, c = idx % 192;
        CUR[m * LD_V + c] = gelu_f(CUR[m * LD_V + c]); }
    __syncthreads();
    GEMM(192, 192, OFF_E2, e2w, e2b, CUR, LD_V, Q, LD_V, 0, 192, BIGM, 1, 0);
    __syncthreads();
    layernorm_rows(Q, LD_V, 192, n2g, n2b);
    __syncthreads();
    FORALL(idx, TILE_M * 192) { int m = idx / 192, c = idx % 192;
        Q[m * LD_V + c] = gelu_f(Q[m * LD_V + c]); }
    __syncthreads();

    // ---- Stage 3: multi-hop memory
    FORALL(idx, TILE_M * 192) { int m = idx / 192, c = idx % 192;
        CUR[m * LD_V + c] = Q[m * LD_V + c]; TOT[m * LD_V + c] = 0.f; }
    __syncthreads();
    for (int h = 0; h < 3; ++h) {
        GEMM(384, 192, OFF_MEMK + h * pkhalves(384, 192),
             mem_keys + h * 384 * 192, nullptr, CUR, LD_V, SC, LD_SC, 0, 192, BIGM, 1, 0);
        __syncthreads();
        softmax_rows(SC, LD_SC, 384, RSCALE);
        __syncthreads();
        GEMM(192, 384, OFF_MEMV + h * pkhalves(192, 384),
             mem_vals + h * 384 * 192, nullptr, SC, LD_SC, R, LD_V, 0, 1, BIGM, 192, 0);
        __syncthreads();
        FORALL(idx, TILE_M * 192) { int m = idx / 192, c = idx % 192;
            TOT[m * LD_V + c] += R[m * LD_V + c]; }
        __syncthreads();
        if (h < 2) {
            FORALL(idx, TILE_M * 384) { int m = idx / 384, c = idx % 384;
                CAT[m * LD_CAT + c] = (c < 192) ? CUR[m * LD_V + c]
                                                : R[m * LD_V + (c - 192)]; }
            __syncthreads();
            const float* rw = h ? rl1w : rl0w; const float* rb = h ? rl1b : rl0b;
            const float* rg = h ? rn1g : rn0g; const float* rnb = h ? rn1b : rn0b;
            GEMM(192, 384, (h ? OFF_RL1 : OFF_RL0), rw, rb, CAT, LD_CAT, DIR, LD_V, 0,
                 384, BIGM, 1, 0);
            __syncthreads();
            layernorm_rows(DIR, LD_V, 192, rg, rnb);
            __syncthreads();
            FORALL(idx, TILE_M * 192) { int m = idx / 192, c = idx % 192;
                CUR[m * LD_V + c] = gelu_f(DIR[m * LD_V + c]) + Q[m * LD_V + c]; }
            __syncthreads();
        }
    }

    // ---- Stage 4: hopfield one-step completion
    GEMM(192, 192, OFF_HQ, hqw, hqb, Q, LD_V, R, LD_V, 0, 192, BIGM, 1, 0);
    __syncthreads();
    GEMM(256, 192, OFF_HPT, hop_pat, nullptr, R, LD_V, SC, LD_SC, 0, 192, BIGM, 1, 0);
    __syncthreads();
    softmax_rows(SC, LD_SC, 256, RSCALE);
    __syncthreads();
    GEMM(192, 256, OFF_HPP, hop_pat, nullptr, SC, LD_SC, R, LD_V, 0, 1, BIGM, 192, 0);
    __syncthreads();
    GEMM(192, 192, OFF_HV, hvw, hvb, R, LD_V, HOP, LD_V, 0, 192, BIGM, 1, 0);
    __syncthreads();
    layernorm_rows(HOP, LD_V, 192, hng, hnb);
    __syncthreads();

    // ---- Stage 5: direct memory
    GEMM(384, 192, OFF_DMK, dmk, nullptr, Q, LD_V, SC, LD_SC, 0, 192, BIGM, 1, 0);
    __syncthreads();
    softmax_rows(SC, LD_SC, 384, RSCALE);
    __syncthreads();
    GEMM(192, 384, OFF_DMV, dmv, nullptr, SC, LD_SC, DIR, LD_V, 0, 1, BIGM, 192, 0);
    __syncthreads();

    // ---- Stage 6: gated fusion
    FORALL(idx, TILE_M * 576) { int m = idx / 576, c = idx % 576;
        float v = (c < 192) ? TOT[m * LD_V + c]
                : (c < 384) ? HOP[m * LD_V + (c - 192)]
                            : DIR[m * LD_V + (c - 384)];
        CAT[m * LD_CAT + c] = v; }
    __syncthreads();
    if (threadIdx.x < 48) {
        int m = threadIdx.x / 3, j = threadIdx.x % 3;
        float s = gb[j];
        for (int k = 0; k < 576; ++k) s += CAT[m * LD_CAT + k] * gw[j * 576 + k];
        GT[m * 4 + j] = 1.f / (1.f + __expf(-s));
    }
    __syncthreads();
    FORALL(idx, TILE_M * 576) { int m = idx / 576, c = idx % 576;
        CAT[m * LD_CAT + c] *= GT[m * 4 + c / 192]; }
    __syncthreads();
    GEMM(192, 576, OFF_FUSE, fw, fb, CAT, LD_CAT, R, LD_V, 0, 576, BIGM, 1, 0);
    __syncthreads();

    // ---- Stage 7: head
    FORALL(idx, TILE_M * 256) { int m = idx / 256, c = idx % 256;
        float v;
        if      (c < 192) v = R[m * LD_V + c];
        else if (c < 208) v = bit_embed[bit_pos[b0 + m] * 16 + (c - 192)];
        else if (c < 215) v = partial[(b0 + m) * 7 + (c - 208)];
        else if (c < 247) v = instinct[(b0 + m) * 32 + (c - 215)];
        else              v = 0.f;
        CAT[m * LD_CAT + c] = v; }
    __syncthreads();
    GEMM(384, 247, OFF_H1, h1w, h1b, CAT, LD_CAT, SC, LD_SC, 0, 247, BIGM, 1, 0);
    __syncthreads();
    layernorm_rows(SC, LD_SC, 384, hn1g, hn1b);
    __syncthreads();
    FORALL(idx, TILE_M * 384) { int m = idx / 384, c = idx % 384;
        SC[m * LD_SC + c] = gelu_f(SC[m * LD_SC + c]); }
    __syncthreads();
    GEMM(192, 384, OFF_H2, h2w, h2b, SC, LD_SC, DIR, LD_V, 0, 384, BIGM, 1, 0);
    __syncthreads();
    FORALL(idx, TILE_M * 192) { int m = idx / 192, c = idx % 192;
        DIR[m * LD_V + c] = gelu_f(DIR[m * LD_V + c]); }
    __syncthreads();
    if (threadIdx.x < TILE_M) {
        int m = threadIdx.x;
        float s = h3b[0];
        for (int k = 0; k < 192; ++k) s += DIR[m * LD_V + k] * h3w[k];
        out[b0 + m] = s;
    }
}

extern "C" void kernel_launch(void* const* d_in, const int* in_sizes, int n_in,
                              void* d_out, int out_size, void* d_ws, size_t ws_size,
                              hipStream_t stream) {
    const int*   ctx  = (const int*)  d_in[0];
    const int*   bitp = (const int*)  d_in[1];
    const float* part = (const float*)d_in[2];
    const float* inst = (const float*)d_in[3];
    auto F = [&](int i) { return (const float*)d_in[i]; };

    const bool packed = ws_size >= (size_t)PK_TOTAL * sizeof(_Float16);
    _Float16* PKW = (_Float16*)d_ws;

    if (packed) {
        auto PKL = [&](int off, const float* W, int N, int Ktot,
                       int ldn, int KM, int S1, int S2) {
            int Kpad = (Ktot + 31) & ~31;
            int total = N * Kpad;
            hipLaunchKernelGGL(pack_w, dim3((total + 255) / 256), dim3(256), 0, stream,
                               W, PKW + off, N, Ktot, ldn, KM, S1, S2);
        };
        PKL(OFF_C2,   F(6),  96,  384, 384, 192, 2, 1);   // conv2 taps
        PKL(OFF_C3,   F(8),  96,  384, 576, 192, 3, 1);   // conv3 taps 0,1
        PKL(OFF_C4,   F(10), 96,  576, 768, 192, 4, 1);   // conv4 taps 0..2
        PKL(OFF_PDP,  F(14), 192, 288, 288, BIGM, 1, 0);
        PKL(OFF_E1,   F(16), 192, 384, 384, BIGM, 1, 0);
        PKL(OFF_E2,   F(20), 192, 192, 192, BIGM, 1, 0);
        PKL(OFF_MEMK, F(24), 1152, 192, 192, BIGM, 1, 0); // 3 hops at once
        for (int h = 0; h < 3; ++h)
            PKL(OFF_MEMV + h * pkhalves(192, 384), F(25) + h * 384 * 192,
                192, 384, 1, BIGM, 192, 0);               // plain V[k][n]
        PKL(OFF_RL0,  F(26), 192, 384, 384, BIGM, 1, 0);
        PKL(OFF_RL1,  F(28), 192, 384, 384, BIGM, 1, 0);
        PKL(OFF_HQ,   F(35), 192, 192, 192, BIGM, 1, 0);
        PKL(OFF_HPT,  F(34), 256, 192, 192, BIGM, 1, 0);  // patterns^T
        PKL(OFF_HPP,  F(34), 192, 256, 1,   BIGM, 192, 0);// patterns plain
        PKL(OFF_HV,   F(37), 192, 192, 192, BIGM, 1, 0);
        PKL(OFF_DMK,  F(41), 384, 192, 192, BIGM, 1, 0);
        PKL(OFF_DMV,  F(42), 192, 384, 1,   BIGM, 192, 0);
        PKL(OFF_FUSE, F(45), 192, 576, 576, BIGM, 1, 0);
        PKL(OFF_H1,   F(48), 384, 247, 247, BIGM, 1, 0);
        PKL(OFF_H2,   F(52), 192, 384, 384, BIGM, 1, 0);
    }

    dim3 grid(4096 / TILE_M), block(NTHR);
    if (packed) {
        hipLaunchKernelGGL(abgpt_fused<true>, grid, block, 0, stream,
            ctx, bitp, part, inst,
            F(4),  F(5),  F(6),  F(7),  F(8),  F(9),  F(10), F(11),
            F(12), F(13), F(14), F(15), F(16), F(17), F(18), F(19),
            F(20), F(21), F(22), F(23), F(24), F(25),
            F(26), F(27), F(28), F(29), F(30), F(31), F(32), F(33),
            F(34), F(35), F(36), F(37), F(38), F(39), F(40),
            F(41), F(42), F(43), F(44), F(45), F(46), F(47),
            F(48), F(49), F(50), F(51), F(52), F(53), F(54), F(55),
            (const _Float16*)PKW, (float*)d_out);
    } else {
        hipLaunchKernelGGL(abgpt_fused<false>, grid, block, 0, stream,
            ctx, bitp, part, inst,
            F(4),  F(5),  F(6),  F(7),  F(8),  F(9),  F(10), F(11),
            F(12), F(13), F(14), F(15), F(16), F(17), F(18), F(19),
            F(20), F(21), F(22), F(23), F(24), F(25),
            F(26), F(27), F(28), F(29), F(30), F(31), F(32), F(33),
            F(34), F(35), F(36), F(37), F(38), F(39), F(40),
            F(41), F(42), F(43), F(44), F(45), F(46), F(47),
            F(48), F(49), F(50), F(51), F(52), F(53), F(54), F(55),
            (const _Float16*)nullptr, (float*)d_out);
    }
}
